// BoundaryRefinementModule_50981261804144
// MI455X (gfx1250) — matrix operand order repaired and checked by reading, not compiled, not run on hardware
//
#include <hip/hip_runtime.h>
#include <cmath>

#define BB 8
#define CCH 256
#define HH 128
#define WW 128
#define SS (HH*WW)

typedef __bf16 bf16x16 __attribute__((ext_vector_type(16)));
typedef float f32x8 __attribute__((ext_vector_type(8)));
typedef int v4i __attribute__((vector_size(4 * sizeof(int))));
typedef __attribute__((address_space(1))) v4i* g4p;
typedef __attribute__((address_space(3))) v4i* l4p;

#if defined(__has_builtin)
#if __has_builtin(__builtin_amdgcn_global_load_async_to_lds_b128) && \
    __has_builtin(__builtin_amdgcn_s_wait_asynccnt)
#define HAVE_ASYNC_LDS 1
#endif
#endif

__device__ __forceinline__ unsigned short f2bf(float f) {
  union { float f; unsigned u; } c; c.f = f;
  unsigned u = c.u;
  unsigned r = (u + 0x7FFFu + ((u >> 16) & 1u)) >> 16;
  return (unsigned short)r;
}
__device__ __forceinline__ float bf2f(unsigned short h) {
  union { unsigned u; float f; } c; c.u = ((unsigned)h) << 16;
  return c.f;
}
__device__ __forceinline__ float geluf(float x) {
  return 0.5f * x * (1.0f + erff(x * 0.70710678118654752440f));
}

// ---------------------------------------------------------------------------
// Weight convert: [Cout][CinS][3][3] f32 -> [9][Cout][CinD] bf16, zero padded
// ---------------------------------------------------------------------------
__global__ void k_wconv(const float* __restrict__ src, unsigned short* __restrict__ dst,
                        int Cout, int CinS, int CinD) {
  int i = blockIdx.x * 256 + threadIdx.x;
  int total = 9 * Cout * CinD;
  if (i >= total) return;
  int ci = i % CinD;
  int co = (i / CinD) % Cout;
  int t  = i / (Cout * CinD);
  float v = (ci < CinS) ? src[((size_t)co * CinS + ci) * 9 + t] : 0.0f;
  dst[i] = f2bf(v);
}

// ---------------------------------------------------------------------------
// Channel mean over 256 channels
// ---------------------------------------------------------------------------
__global__ __launch_bounds__(256) void k_mean_c(const float* __restrict__ fused,
                                                float* __restrict__ xm) {
  int i = blockIdx.x * 256 + threadIdx.x;
  if (i >= BB * SS) return;
  int b = i / SS, p = i % SS;
  const float* base = fused + (size_t)b * CCH * SS + p;
  float s = 0.0f;
  for (int c = 0; c < CCH; ++c) s += base[(size_t)c * SS];
  xm[i] = s * (1.0f / (float)CCH);
}

// ---------------------------------------------------------------------------
// Sobel magnitude + per-block min/max partials
// ---------------------------------------------------------------------------
__global__ __launch_bounds__(256) void k_sobel(const float* __restrict__ xm,
                                               float* __restrict__ emag,
                                               float* __restrict__ partials) {
  int b = blockIdx.y;
  int p = blockIdx.x * 256 + threadIdx.x;
  int y = p / WW, x = p % WW;
  const float* m = xm + (size_t)b * SS;
  float a00 = (y > 0 && x > 0)            ? m[(y-1)*WW + (x-1)] : 0.0f;
  float a01 = (y > 0)                     ? m[(y-1)*WW + x]     : 0.0f;
  float a02 = (y > 0 && x < WW-1)         ? m[(y-1)*WW + (x+1)] : 0.0f;
  float a10 = (x > 0)                     ? m[y*WW + (x-1)]     : 0.0f;
  float a12 = (x < WW-1)                  ? m[y*WW + (x+1)]     : 0.0f;
  float a20 = (y < HH-1 && x > 0)         ? m[(y+1)*WW + (x-1)] : 0.0f;
  float a21 = (y < HH-1)                  ? m[(y+1)*WW + x]     : 0.0f;
  float a22 = (y < HH-1 && x < WW-1)      ? m[(y+1)*WW + (x+1)] : 0.0f;
  float gx = -a00 + a02 - 2.0f*a10 + 2.0f*a12 - a20 + a22;
  float gy = -a00 - 2.0f*a01 - a02 + a20 + 2.0f*a21 + a22;
  float mg = sqrtf(gx*gx + gy*gy + 1e-8f);
  emag[(size_t)b * SS + p] = mg;

  __shared__ float smin[256], smax[256];
  smin[threadIdx.x] = mg; smax[threadIdx.x] = mg;
  __syncthreads();
  for (int o = 128; o > 0; o >>= 1) {
    if (threadIdx.x < (unsigned)o) {
      smin[threadIdx.x] = fminf(smin[threadIdx.x], smin[threadIdx.x + o]);
      smax[threadIdx.x] = fmaxf(smax[threadIdx.x], smax[threadIdx.x + o]);
    }
    __syncthreads();
  }
  if (threadIdx.x == 0) {
    partials[(b * gridDim.x + blockIdx.x) * 2 + 0] = smin[0];
    partials[(b * gridDim.x + blockIdx.x) * 2 + 1] = smax[0];
  }
}

__global__ __launch_bounds__(64) void k_minmax_final(const float* __restrict__ partials,
                                                     int nb, float* __restrict__ mm) {
  int b = blockIdx.x;
  __shared__ float smin[64], smax[64];
  float mn = 3.4e38f, mx = -3.4e38f;
  for (int i = threadIdx.x; i < nb; i += 64) {
    mn = fminf(mn, partials[(b * nb + i) * 2 + 0]);
    mx = fmaxf(mx, partials[(b * nb + i) * 2 + 1]);
  }
  smin[threadIdx.x] = mn; smax[threadIdx.x] = mx;
  __syncthreads();
  for (int o = 32; o > 0; o >>= 1) {
    if (threadIdx.x < (unsigned)o) {
      smin[threadIdx.x] = fminf(smin[threadIdx.x], smin[threadIdx.x + o]);
      smax[threadIdx.x] = fmaxf(smax[threadIdx.x], smax[threadIdx.x + o]);
    }
    __syncthreads();
  }
  if (threadIdx.x == 0) { mm[b*2] = smin[0]; mm[b*2+1] = smax[0]; }
}

__global__ void k_edge_norm(float* __restrict__ edge, const float* __restrict__ mm) {
  int i = blockIdx.x * 256 + threadIdx.x;
  if (i >= BB * SS) return;
  int b = i / SS;
  edge[i] = (edge[i] - mm[b*2]) / (mm[b*2+1] - mm[b*2] + 1e-8f);
}

// ---------------------------------------------------------------------------
// Depthwise Laplacian -> bf16
// ---------------------------------------------------------------------------
__global__ void k_laplacian(const float* __restrict__ fused, unsigned short* __restrict__ hf) {
  size_t i = (size_t)blockIdx.x * 256 + threadIdx.x;
  if (i >= (size_t)BB * CCH * SS) return;
  int p = (int)(i % SS);
  int y = p / WW, x = p % WW;
  const float* base = fused + (i - p);
  float c  = base[p];
  float up = (y > 0)      ? base[p - WW] : 0.0f;
  float dn = (y < HH - 1) ? base[p + WW] : 0.0f;
  float lf = (x > 0)      ? base[p - 1]  : 0.0f;
  float rt = (x < WW - 1) ? base[p + 1]  : 0.0f;
  hf[i] = f2bf(up + dn + lf + rt - 4.0f * c);
}

// ---------------------------------------------------------------------------
// Implicit-GEMM 3x3 SAME conv via V_WMMA_F32_16X16X32_BF16.
// Block = 256 threads (8 waves). Tile = 64 Cout x 128 pixels (one full row).
// Each wave: 1 M-frag x 4 N-frags (4 f32x8 accumulators).
// in:  bf16 [B][Cin][H][W], w: bf16 [9][Cout][Cin].
// mode 0: out_f32 = acc
// mode 1: out_bf16 = bf16(fused + edge*acc)          (ref_in fusion)
// mode 2: out_f32 = fused + rs*gate*acc              (final refined)
// ---------------------------------------------------------------------------
__global__ __launch_bounds__(256)
void k_conv3x3_wmma(const unsigned short* __restrict__ in, int Cin,
                    const unsigned short* __restrict__ w, int Cout,
                    float* __restrict__ out_f32,
                    unsigned short* __restrict__ out_bf16,
                    int mode,
                    const float* __restrict__ fused,
                    const float* __restrict__ edge,
                    const float* __restrict__ gate,
                    const float* __restrict__ rs_ptr) {
  __shared__ unsigned short sX[3 * 130 * 32];  // [row][x-1..x+128][cblk] transposed
  __shared__ unsigned short sW[9 * 64 * 32];   // [tap][m][k]

  int co0 = blockIdx.x * 64;
  int y   = blockIdx.y;
  int b   = blockIdx.z;

  int tid  = threadIdx.x;
  int lane = tid & 31;
  int wv   = tid >> 5;
  int mi   = wv & 3;    // M fragment row (0..3)
  int nh   = wv >> 2;   // N half (0..1), 4 frags of 16 each
  int lrow = lane & 15;
  int hi   = (lane >> 4) & 1;

  f32x8 acc[4] = {};

  const unsigned short* inB = in + (size_t)b * Cin * SS;

  for (int c0 = 0; c0 < Cin; c0 += 32) {
    __syncthreads();
    // Stage weights: 9 taps x 64 co x 32 ci from [tap][Cout][Cin].
#if defined(HAVE_ASYNC_LDS)
    // Contiguous on both sides -> async LDS DMA (ASYNCcnt), 16B per lane op.
    for (int i = tid; i < (9 * 64 * 32) / 8; i += 256) {
      int h0 = i * 8;
      int k  = h0 & 31;
      int m  = (h0 >> 5) & 63;
      int t  = h0 >> 11;
      const unsigned short* gp = w + ((size_t)t * Cout + (co0 + m)) * Cin + (c0 + k);
      __builtin_amdgcn_global_load_async_to_lds_b128(
          (g4p)(__UINTPTR_TYPE__)(gp),
          (l4p)(unsigned int)(__UINTPTR_TYPE__)(&sW[h0]),
          0, 0);
    }
#else
    for (int i = tid; i < 9 * 64 * 32; i += 256) {
      int k = i & 31;
      int m = (i >> 5) & 63;
      int t = i >> 11;
      sW[i] = w[((size_t)t * Cout + (co0 + m)) * Cin + (c0 + k)];
    }
#endif
    // Stage input strip: 3 halo rows x 130 x 32 channels (transposed [x][c])
    for (int i = tid; i < 3 * 130 * 32; i += 256) {
      int k  = i & 31;
      int t  = i >> 5;
      int xx = t % 130;
      int r  = t / 130;
      int yy = y + r - 1;
      int gx = xx - 1;
      unsigned short v = 0;
      if (yy >= 0 && yy < HH && gx >= 0 && gx < WW)
        v = inB[(size_t)(c0 + k) * SS + yy * WW + gx];
      sX[(r * 130 + xx) * 32 + k] = v;
    }
    // Prefetch next K-block input rows into cache while we compute.
    if (c0 + 32 < Cin) {
      int k = tid & 31;
      int r = (tid >> 5) & 3;
      if (r < 3) {
        int yy = y + r - 1;
        if (yy >= 0 && yy < HH)
          __builtin_prefetch(&inB[(size_t)(c0 + 32 + k) * SS + yy * WW], 0, 1);
      }
    }
#if defined(HAVE_ASYNC_LDS)
    __builtin_amdgcn_s_wait_asynccnt(0);
#endif
    __syncthreads();

    for (int ky = 0; ky < 3; ++ky) {
      const unsigned short* sXr = &sX[ky * 130 * 32];
      #pragma unroll
      for (int kx = 0; kx < 3; ++kx) {
        // A fragment (ISA 16x32 bf16 layout)
        union { bf16x16 v; unsigned u[8]; } A;
        const unsigned short* sWt = &sW[((ky * 3 + kx) * 64 + (mi * 16 + lrow)) * 32];
        #pragma unroll
        for (int p = 0; p < 8; ++p) {
          int kk = (p < 4 ? 2 * p : 2 * p + 8) + (hi ? 8 : 0);
          A.u[p] = *(const unsigned*)&sWt[kk];
        }
        #pragma unroll
        for (int f = 0; f < 4; ++f) {
          int nn = (nh * 4 + f) * 16 + lrow;
          const unsigned short* sXp = &sXr[(nn + kx) * 32];
          union { bf16x16 v; unsigned u[8]; } Bf;
          #pragma unroll
          for (int p = 0; p < 8; ++p) Bf.u[p] = *(const unsigned*)&sXp[(p < 4 ? 2 * p : 2 * p + 8) + (hi ? 8 : 0)];
          acc[f] = __builtin_amdgcn_wmma_f32_16x16x32_bf16(false, A.v, false, Bf.v,
                                                           (short)0, acc[f], false, false);
        }
      }
    }
  }

  float rs = (mode == 2) ? rs_ptr[0] : 0.0f;
  #pragma unroll
  for (int f = 0; f < 4; ++f) {
    int ncol = (nh * 4 + f) * 16 + lrow;
    #pragma unroll
    for (int r = 0; r < 8; ++r) {
      int m  = r + (hi ? 8 : 0);
      int co = co0 + mi * 16 + m;
      size_t oi = (((size_t)b * Cout + co) * HH + y) * WW + ncol;
      float a = acc[f][r];
      if (mode == 0) {
        out_f32[oi] = a;
      } else if (mode == 1) {
        size_t fi = (((size_t)b * CCH + co) * HH + y) * WW + ncol;
        float e = edge[(size_t)b * SS + y * WW + ncol];
        out_bf16[oi] = f2bf(fused[fi] + e * a);
      } else {
        size_t fi = (((size_t)b * CCH + co) * HH + y) * WW + ncol;
        float gt = gate[(size_t)b * SS + y * WW + ncol];
        out_f32[oi] = fused[fi] + rs * gt * a;
      }
    }
  }
}

// ---------------------------------------------------------------------------
// GroupNorm stats (8 groups): one block per (b, group)
// ---------------------------------------------------------------------------
__global__ __launch_bounds__(256) void k_gn_stats(const float* __restrict__ x, int Cc,
                                                  float* __restrict__ stats) {
  int bg = blockIdx.x;
  int b = bg >> 3, g = bg & 7;
  int cpg = Cc / 8;
  size_t n = (size_t)cpg * SS;
  const float* base = x + ((size_t)b * Cc + g * cpg) * SS;
  float s = 0.0f, sq = 0.0f;
  for (size_t i = threadIdx.x; i < n; i += 256) {
    float v = base[i];
    s += v; sq += v * v;
  }
  __shared__ float rs_[256], rq_[256];
  rs_[threadIdx.x] = s; rq_[threadIdx.x] = sq;
  __syncthreads();
  for (int o = 128; o > 0; o >>= 1) {
    if (threadIdx.x < (unsigned)o) {
      rs_[threadIdx.x] += rs_[threadIdx.x + o];
      rq_[threadIdx.x] += rq_[threadIdx.x + o];
    }
    __syncthreads();
  }
  if (threadIdx.x == 0) {
    float mean = rs_[0] / (float)n;
    float var  = rq_[0] / (float)n - mean * mean;
    stats[bg * 2 + 0] = mean;
    stats[bg * 2 + 1] = rsqrtf(var + 1e-5f);
  }
}

__global__ void k_gn_act(const float* __restrict__ x, int Cc,
                         const float* __restrict__ stats,
                         const float* __restrict__ gw, const float* __restrict__ gb,
                         unsigned short* __restrict__ out) {
  size_t i = (size_t)blockIdx.x * 256 + threadIdx.x;
  size_t total = (size_t)BB * Cc * SS;
  if (i >= total) return;
  int c = (int)((i / SS) % Cc);
  int b = (int)(i / ((size_t)Cc * SS));
  int g = c / (Cc / 8);
  float mean = stats[(b * 8 + g) * 2 + 0];
  float rstd = stats[(b * 8 + g) * 2 + 1];
  float v = (x[i] - mean) * rstd * gw[c] + gb[c];
  out[i] = f2bf(geluf(v));
}

// ---------------------------------------------------------------------------
// Gate input concat: [fused(256) | edge(1) | zeros(31)] -> bf16 288ch
// ---------------------------------------------------------------------------
__global__ void k_gate_in(const float* __restrict__ fused, const float* __restrict__ edge,
                          unsigned short* __restrict__ gi) {
  size_t i = (size_t)blockIdx.x * 256 + threadIdx.x;
  size_t total = (size_t)BB * 288 * SS;
  if (i >= total) return;
  int p = (int)(i % SS);
  int c = (int)((i / SS) % 288);
  int b = (int)(i / ((size_t)288 * SS));
  float v;
  if (c < 256)       v = fused[((size_t)b * CCH + c) * SS + p];
  else if (c == 256) v = edge[(size_t)b * SS + p];
  else               v = 0.0f;
  gi[i] = f2bf(v);
}

// ---------------------------------------------------------------------------
// Gate 1x1 conv (64 -> 1) + sigmoid + per-block mean partials
// ---------------------------------------------------------------------------
__global__ __launch_bounds__(256) void k_gate1x1(const unsigned short* __restrict__ g1a,
                                                 const float* __restrict__ w2,
                                                 const float* __restrict__ b2,
                                                 float* __restrict__ gate,
                                                 float* __restrict__ partials) {
  int b = blockIdx.y;
  int p = blockIdx.x * 256 + threadIdx.x;
  const unsigned short* base = g1a + (size_t)b * 64 * SS + p;
  float s = 0.0f;
  for (int c = 0; c < 64; ++c) s += bf2f(base[(size_t)c * SS]) * w2[c];
  float gt = 1.0f / (1.0f + expf(-(s + b2[0])));
  gate[(size_t)b * SS + p] = gt;
  __shared__ float sh[256];
  sh[threadIdx.x] = gt;
  __syncthreads();
  for (int o = 128; o > 0; o >>= 1) {
    if (threadIdx.x < (unsigned)o) sh[threadIdx.x] += sh[threadIdx.x + o];
    __syncthreads();
  }
  if (threadIdx.x == 0) partials[b * gridDim.x + blockIdx.x] = sh[0];
}

__global__ __launch_bounds__(256) void k_gate_mean(const float* __restrict__ partials,
                                                   int n, float* __restrict__ out) {
  __shared__ float sh[256];
  float s = 0.0f;
  for (int i = threadIdx.x; i < n; i += 256) s += partials[i];
  sh[threadIdx.x] = s;
  __syncthreads();
  for (int o = 128; o > 0; o >>= 1) {
    if (threadIdx.x < (unsigned)o) sh[threadIdx.x] += sh[threadIdx.x + o];
    __syncthreads();
  }
  if (threadIdx.x == 0) out[0] = sh[0] / (float)((size_t)BB * SS);
}

// ---------------------------------------------------------------------------
// Host orchestration.  Requires ws_size >= ~240 MB (peak-aliased arena).
// ---------------------------------------------------------------------------
extern "C" void kernel_launch(void* const* d_in, const int* in_sizes, int n_in,
                              void* d_out, int out_size, void* d_ws, size_t ws_size,
                              hipStream_t stream) {
  (void)in_sizes; (void)n_in; (void)out_size; (void)ws_size;
  const float* fused   = (const float*)d_in[0];
  const float* hf_w1   = (const float*)d_in[1];
  const float* hf_gnw  = (const float*)d_in[2];
  const float* hf_gnb  = (const float*)d_in[3];
  const float* hf_w2   = (const float*)d_in[4];
  const float* gate_w1 = (const float*)d_in[5];
  const float* gt_gnw  = (const float*)d_in[6];
  const float* gt_gnb  = (const float*)d_in[7];
  const float* gate_w2 = (const float*)d_in[8];
  const float* gate_b2 = (const float*)d_in[9];
  const float* ref_w1  = (const float*)d_in[10];
  const float* ref_gnw = (const float*)d_in[11];
  const float* ref_gnb = (const float*)d_in[12];
  const float* ref_w2  = (const float*)d_in[13];
  const float* rs_ptr  = (const float*)d_in[14];

  char* ws = (char*)d_ws;
  float* out = (float*)d_out;
  float* edge_out  = out + (size_t)BB * CCH * SS;              // B*S edge map (output)
  float* gmean_out = out + (size_t)BB * CCH * SS + (size_t)BB * SS;

  // Small-region offsets
  float* xm       = (float*)(ws + 0x000000);                   // 512 KB
  float* mm       = (float*)(ws + 0x080000);                   // min/max per batch
  float* gnstats  = (float*)(ws + 0x080400);
  float* partials = (float*)(ws + 0x081000);                   // sobel/gate partials
  float* gatebuf  = (float*)(ws + 0x090000);                   // 512 KB gate map
  unsigned short* wb = (unsigned short*)(ws + 0x110000);       // bf16 weights arena
  unsigned short* wb_hf1  = wb;                                // 9*128*256
  unsigned short* wb_hf2  = wb + 294912;                       // 9*256*128
  unsigned short* wb_gate = wb + 589824;                       // 9*64*288
  unsigned short* wb_ref1 = wb + 755712;                       // 9*256*256
  unsigned short* wb_ref2 = wb + 1345536;                      // 9*256*256

  // Big arena (lifetimes staged; aliasing is intentional and safe in order)
  char* AR = ws + 0x800000;
  unsigned short* hf_bf   = (unsigned short*)(AR + 0x0000000); // 64 MB
  float*          h1_f32  = (float*)        (AR + 0x4000000);  // 64 MB
  unsigned short* h1a_bf  = (unsigned short*)(AR + 0x8000000); // 32 MB
  unsigned short* refin   = (unsigned short*)(AR + 0xA000000); // 64 MB
  unsigned short* gatein  = (unsigned short*)(AR + 0x0000000); // 72 MB (aliases hf,h1)
  float*          g1_f32  = (float*)        (AR + 0x8000000);  // 32 MB (aliases h1a)
  unsigned short* g1a_bf  = (unsigned short*)(AR + 0x4800000); // 16 MB
  float*          r1_f32  = (float*)        (AR + 0x0000000);  // 128 MB
  unsigned short* r1a_bf  = (unsigned short*)(AR + 0x8000000); // 64 MB (aliases refin head)

  dim3 blk(256);

  // 1. Weights -> bf16 [tap][co][ci] (gate padded 257->288)
  k_wconv<<<dim3((9*128*256 + 255)/256), blk, 0, stream>>>(hf_w1,   wb_hf1,  128, 256, 256);
  k_wconv<<<dim3((9*256*128 + 255)/256), blk, 0, stream>>>(hf_w2,   wb_hf2,  256, 128, 128);
  k_wconv<<<dim3((9*64*288  + 255)/256), blk, 0, stream>>>(gate_w1, wb_gate,  64, 257, 288);
  k_wconv<<<dim3((9*256*256 + 255)/256), blk, 0, stream>>>(ref_w1,  wb_ref1, 256, 256, 256);
  k_wconv<<<dim3((9*256*256 + 255)/256), blk, 0, stream>>>(ref_w2,  wb_ref2, 256, 256, 256);

  // 2. Edge map: channel mean, Sobel, per-batch min/max, normalize (into d_out)
  k_mean_c<<<dim3((BB*SS)/256), blk, 0, stream>>>(fused, xm);
  k_sobel<<<dim3(SS/256, BB), blk, 0, stream>>>(xm, edge_out, partials);
  k_minmax_final<<<dim3(BB), dim3(64), 0, stream>>>(partials, SS/256, mm);
  k_edge_norm<<<dim3((BB*SS)/256), blk, 0, stream>>>(edge_out, mm);

  // 3. High-frequency path
  k_laplacian<<<dim3((unsigned)(((size_t)BB*CCH*SS)/256)), blk, 0, stream>>>(fused, hf_bf);
  k_conv3x3_wmma<<<dim3(128/64, HH, BB), blk, 0, stream>>>(
      hf_bf, 256, wb_hf1, 128, h1_f32, nullptr, 0, nullptr, nullptr, nullptr, nullptr);
  k_gn_stats<<<dim3(BB*8), blk, 0, stream>>>(h1_f32, 128, gnstats);
  k_gn_act<<<dim3((unsigned)(((size_t)BB*128*SS)/256)), blk, 0, stream>>>(
      h1_f32, 128, gnstats, hf_gnw, hf_gnb, h1a_bf);
  // conv hf_w2 fused with ref_in = bf16(fused + edge * hf_features)
  k_conv3x3_wmma<<<dim3(256/64, HH, BB), blk, 0, stream>>>(
      h1a_bf, 128, wb_hf2, 256, nullptr, refin, 1, fused, edge_out, nullptr, nullptr);

  // 4. Boundary-aware gate
  k_gate_in<<<dim3((unsigned)(((size_t)BB*288*SS)/256)), blk, 0, stream>>>(fused, edge_out, gatein);
  k_conv3x3_wmma<<<dim3(64/64, HH, BB), blk, 0, stream>>>(
      gatein, 288, wb_gate, 64, g1_f32, nullptr, 0, nullptr, nullptr, nullptr, nullptr);
  k_gn_stats<<<dim3(BB*8), blk, 0, stream>>>(g1_f32, 64, gnstats);
  k_gn_act<<<dim3((unsigned)(((size_t)BB*64*SS)/256)), blk, 0, stream>>>(
      g1_f32, 64, gnstats, gt_gnw, gt_gnb, g1a_bf);
  k_gate1x1<<<dim3(SS/256, BB), blk, 0, stream>>>(g1a_bf, gate_w2, gate_b2, gatebuf, partials);
  k_gate_mean<<<dim3(1), blk, 0, stream>>>(partials, (SS/256)*BB, gmean_out);

  // 5. Refine path
  k_conv3x3_wmma<<<dim3(256/64, HH, BB), blk, 0, stream>>>(
      refin, 256, wb_ref1, 256, r1_f32, nullptr, 0, nullptr, nullptr, nullptr, nullptr);
  k_gn_stats<<<dim3(BB*8), blk, 0, stream>>>(r1_f32, 256, gnstats);
  k_gn_act<<<dim3((unsigned)(((size_t)BB*256*SS)/256)), blk, 0, stream>>>(
      r1_f32, 256, gnstats, ref_gnw, ref_gnb, r1a_bf);
  // Final conv fused with refined = fused + rs * gate * refine_feat -> d_out
  k_conv3x3_wmma<<<dim3(256/64, HH, BB), blk, 0, stream>>>(
      r1a_bf, 256, wb_ref2, 256, out, nullptr, 2, fused, nullptr, gatebuf, rs_ptr);
}
